// GCNEmbedding_3315714753134
// MI455X (gfx1250) — matrix operand order repaired and checked by reading, not compiled, not run on hardware
//
#include <hip/hip_runtime.h>
#include <hip/hip_bf16.h>
#include <math.h>

#define D 64

typedef __attribute__((ext_vector_type(16))) __bf16 v16bf;
typedef __attribute__((ext_vector_type(8)))  float  v8f;

// ---------------- degree / norm ----------------
__global__ void k_init_deg(float* __restrict__ deg, int n) {
    int i = blockIdx.x * blockDim.x + threadIdx.x;
    if (i < n) deg[i] = 1.0f;             // self-loop weight
}

__global__ void k_edge_deg(const int* __restrict__ dst, const float* __restrict__ ew,
                           float* __restrict__ deg, int E) {
    int e = blockIdx.x * blockDim.x + threadIdx.x;
    if (e < E) atomicAdd(&deg[dst[e]], ew[e]);
}

__global__ void k_rsqrt_inplace(float* __restrict__ deg, int n) {
    int i = blockIdx.x * blockDim.x + threadIdx.x;
    if (i < n) deg[i] = rsqrtf(deg[i]);   // deg >= 1 always
}

// ---------------- dense GEMM H = X(N x 64) @ W(64 x 64), WMMA bf16 ----------------
// One wave per 16x16 output tile; block = 4 waves = the 4 N-tiles of one M-tile.
__global__ void k_gemm64_wmma(const float* __restrict__ X, const float* __restrict__ W,
                              float* __restrict__ H, int nrows) {
    int lane = threadIdx.x & 31;
    int wave = threadIdx.x >> 5;
    int tileM = blockIdx.x;               // rows 16*tileM .. +15
    int tileN = wave;                     // cols 16*tileN .. +15
    if (tileM * 16 >= nrows) return;      // wave-uniform: EXEC stays all-ones

    int half = lane >> 4;                 // lane group 0/1 per ISA fragment layout
    int row  = tileM * 16 + (lane & 15);
    int col  = tileN * 16 + (lane & 15);

    v16bf a0, a1, b0, b1;
    const float* xr = X + (size_t)row * D;
    #pragma unroll
    for (int e = 0; e < 16; ++e) {
        // A 16x32 bf16 layout: K = (e/8)*16 + half*8 + (e%8)
        int k = ((e >> 3) << 4) + (half << 3) + (e & 7);
        a0[e] = (__bf16)xr[k];
        a1[e] = (__bf16)xr[32 + k];
    }
    #pragma unroll
    for (int e = 0; e < 16; ++e) {
        // B 32x16 bf16 layout: K = (e/8)*16 + half*8 + (e%8), matching A
        int k = ((e >> 3) << 4) + (half << 3) + (e & 7);
        b0[e] = (__bf16)W[k * D + col];
        b1[e] = (__bf16)W[(k + 32) * D + col];
    }
    v8f c = {};
    c = __builtin_amdgcn_wmma_f32_16x16x32_bf16(false, a0, false, b0, (short)0, c, false, false);
    c = __builtin_amdgcn_wmma_f32_16x16x32_bf16(false, a1, false, b1, (short)0, c, false, false);
    #pragma unroll
    for (int r = 0; r < 8; ++r) {
        int rowOut = tileM * 16 + r + (half << 3);   // C/D: VGPR r -> row r + 8*half
        H[(size_t)rowOut * D + col] = c[r];
    }
}

// ---------------- self-loop init: out[n,f] = h[n,f] * dinv[n]^2 ----------------
__global__ void k_self_init(const float* __restrict__ h, const float* __restrict__ dinv,
                            float* __restrict__ out, int total) {
    int i = blockIdx.x * blockDim.x + threadIdx.x;
    if (i < total) {
        float di = dinv[i >> 6];
        out[i] = h[i] * di * di;
    }
}

// ---------------- edge scatter: out[dst] += h[src] * (dinv[s]*w*dinv[d]) ----------------
// 64 lanes per edge -> coalesced gather + coalesced atomics.
__global__ void k_edge_msg(const int* __restrict__ src, const int* __restrict__ dst,
                           const float* __restrict__ ew, const float* __restrict__ dinv,
                           const float* __restrict__ h, float* __restrict__ out, int E) {
    int t = blockIdx.x * blockDim.x + threadIdx.x;
    int e = t >> 6;
    int f = t & 63;
    if (e < E) {
        int s = src[e], d = dst[e];
        float norm = dinv[s] * ew[e] * dinv[d];
        atomicAdd(&out[(size_t)d * D + f], h[(size_t)s * D + f] * norm);
    }
}

// ---------------- bias + SELU (in place) ----------------
__global__ void k_bias_selu(float* __restrict__ out, const float* __restrict__ b, int total) {
    int i = blockIdx.x * blockDim.x + threadIdx.x;
    if (i < total) {
        const float scale = 1.0507009873554805f, alpha = 1.6732632423543772f;
        float v = out[i] + b[i & 63];
        out[i] = (v > 0.0f) ? scale * v : scale * alpha * (expf(v) - 1.0f);
    }
}

// ---------------- per-node einsum + fused LayerNorm ----------------
// One wave per node: lane owns columns j and j+32. Wave32 shfl reductions.
__global__ void k_einsum_ln(const float* __restrict__ h, const float* __restrict__ W1,
                            const float* __restrict__ b1, const float* __restrict__ gamma,
                            const float* __restrict__ beta, float* __restrict__ t, int n) {
    int lane = threadIdx.x & 31;
    int node = blockIdx.x * (blockDim.x >> 5) + (threadIdx.x >> 5);
    if (node >= n) return;
    const float* hn = h  + (size_t)node * D;
    const float* wn = W1 + (size_t)node * D * D;
    float acc0 = b1[(size_t)node * D + lane];
    float acc1 = b1[(size_t)node * D + lane + 32];
    #pragma unroll 8
    for (int d = 0; d < D; ++d) {
        float hd = hn[d];
        acc0 = fmaf(hd, wn[d * D + lane],      acc0);
        acc1 = fmaf(hd, wn[d * D + lane + 32], acc1);
    }
    float s  = acc0 + acc1;
    float sq = acc0 * acc0 + acc1 * acc1;
    #pragma unroll
    for (int off = 16; off > 0; off >>= 1) {
        s  += __shfl_xor(s,  off, 32);
        sq += __shfl_xor(sq, off, 32);
    }
    float mean = s * (1.0f / D);
    float var  = sq * (1.0f / D) - mean * mean;
    float rstd = rsqrtf(var + 1e-5f);
    t[(size_t)node * D + lane]      = (acc0 - mean) * rstd * gamma[lane]      + beta[lane];
    t[(size_t)node * D + lane + 32] = (acc1 - mean) * rstd * gamma[lane + 32] + beta[lane + 32];
}

// ---------------- per-node einsum (final) ----------------
__global__ void k_einsum(const float* __restrict__ t, const float* __restrict__ W2,
                         const float* __restrict__ b2, float* __restrict__ out, int n) {
    int lane = threadIdx.x & 31;
    int node = blockIdx.x * (blockDim.x >> 5) + (threadIdx.x >> 5);
    if (node >= n) return;
    const float* tn = t  + (size_t)node * D;
    const float* wn = W2 + (size_t)node * D * D;
    float acc0 = b2[(size_t)node * D + lane];
    float acc1 = b2[(size_t)node * D + lane + 32];
    #pragma unroll 8
    for (int d = 0; d < D; ++d) {
        float td = tn[d];
        acc0 = fmaf(td, wn[d * D + lane],      acc0);
        acc1 = fmaf(td, wn[d * D + lane + 32], acc1);
    }
    out[(size_t)node * D + lane]      = acc0;
    out[(size_t)node * D + lane + 32] = acc1;
}

extern "C" void kernel_launch(void* const* d_in, const int* in_sizes, int n_in,
                              void* d_out, int out_size, void* d_ws, size_t ws_size,
                              hipStream_t stream) {
    const float* x     = (const float*)d_in[0];
    const int*   ei    = (const int*)  d_in[1];
    const float* ew    = (const float*)d_in[2];
    const float* Wc1   = (const float*)d_in[3];
    const float* bc1   = (const float*)d_in[4];
    const float* Wc2   = (const float*)d_in[5];
    const float* bc2   = (const float*)d_in[6];
    const float* W1    = (const float*)d_in[7];
    const float* b1    = (const float*)d_in[8];
    const float* W2    = (const float*)d_in[9];
    const float* b2    = (const float*)d_in[10];
    const float* gamma = (const float*)d_in[11];
    const float* beta  = (const float*)d_in[12];

    const int N = in_sizes[0] / D;
    const int E = in_sizes[2];
    const int* src = ei;
    const int* dst = ei + E;

    float* dinv = (float*)d_ws;
    float* bufA = dinv + N;                 // GEMM output / gathered features
    float* bufB = bufA + (size_t)N * D;     // scatter accumulator
    float* outF = (float*)d_out;

    const int total = N * D;
    const int TB = 256;
    dim3 blk(TB);
    int gN   = (N + TB - 1) / TB;
    int gE   = (E + TB - 1) / TB;
    int gTot = (total + TB - 1) / TB;
    int gEdgeMsg = (E * D + TB - 1) / TB;   // 64 lanes per edge
    int gNode = (N + (TB / 32) - 1) / (TB / 32);
    int mtiles = (N + 15) / 16;

    // deg -> dinv
    k_init_deg<<<gN, blk, 0, stream>>>(dinv, N);
    k_edge_deg<<<gE, blk, 0, stream>>>(dst, ew, dinv, E);
    k_rsqrt_inplace<<<gN, blk, 0, stream>>>(dinv, N);

    // ---- GCN layer 1 ----
    k_gemm64_wmma<<<mtiles, dim3(128), 0, stream>>>(x, Wc1, bufA, N);
    k_self_init<<<gTot, blk, 0, stream>>>(bufA, dinv, bufB, total);
    k_edge_msg<<<gEdgeMsg, blk, 0, stream>>>(src, dst, ew, dinv, bufA, bufB, E);
    k_bias_selu<<<gTot, blk, 0, stream>>>(bufB, bc1, total);

    // ---- GCN layer 2 ----
    k_gemm64_wmma<<<mtiles, dim3(128), 0, stream>>>(bufB, Wc2, bufA, N);
    k_self_init<<<gTot, blk, 0, stream>>>(bufA, dinv, bufB, total);
    k_edge_msg<<<gEdgeMsg, blk, 0, stream>>>(src, dst, ew, dinv, bufA, bufB, E);
    k_bias_selu<<<gTot, blk, 0, stream>>>(bufB, bc2, total);

    // ---- per-node transform + LayerNorm, then final per-node transform ----
    k_einsum_ln<<<gNode, blk, 0, stream>>>(bufB, W1, b1, gamma, beta, bufA, N);
    k_einsum<<<gNode, blk, 0, stream>>>(bufA, W2, b2, outF, N);
}
